// MultiHeadAttention_22771916603671
// MI455X (gfx1250) — matrix operand order repaired and checked by reading, not compiled, not run on hardware
//
#include <hip/hip_runtime.h>
#include <hip/hip_bf16.h>
#include <math.h>

typedef __bf16 bf16;
typedef __attribute__((ext_vector_type(16))) __bf16 v16bf;
typedef __attribute__((ext_vector_type(8)))  float  v8f;
typedef __attribute__((ext_vector_type(4)))  unsigned v4u;

union FragBF {
    v16bf v;
    v4u   q[2];
    unsigned int w[8];
    bf16 h[16];
};

#define EMB   768
#define HEADS 4
#define HDIM  192
#define BATCH 4
#define SEQ   2048
#define NQKV  2304
#define MROWS (BATCH * SEQ)   // 8192

#define BM  128
#define BN  128
#define BK  32
#define KT  32

// ---- dynamic-LDS byte layout -------------------------------------------------
// GEMM kernels (double buffered):  per buffer: A 128x(32+8)bf16 = 10240B,
//                                              B 128x(32+8)bf16 = 10240B
#define G_A_OFF 0u
#define G_B_OFF 10240u
#define G_BUF   20480u       // buffer stride; total 40960B
// Attention (double buffered):  K  32x(192+8)bf16 = 12800B
//                               Vt 192x(32+8)bf16 = 15360B
#define ATT_V_OFF 12800u
#define ATT_BUF   28160u     // buffer stride
#define ATT_P_OFF 56320u     // P scratch: 8 waves x 16x(32+8)bf16 = 10240B -> total 66560B

// ---- CDNA5 async global->LDS copy (ASYNCcnt) --------------------------------
__device__ __forceinline__ void async_ld_b128(unsigned lds_off, const void* g) {
    asm volatile("global_load_async_to_lds_b128 %0, %1, off"
                 :: "v"(lds_off), "v"((unsigned long long)(uintptr_t)g)
                 : "memory");
}
#define WAIT_ASYNC(N) asm volatile("s_wait_asynccnt " #N ::: "memory")
#define WAIT_DS0()    asm volatile("s_wait_dscnt 0" ::: "memory")

// ---- fragment loaders (layouts per ISA 7.12.2; 16B-aligned -> ds_load_b128) --
__device__ __forceinline__ void frag_ldA(FragBF& f, const char* p) {
    f.q[0] = *(const v4u*)__builtin_assume_aligned(p, 16);
    f.q[1] = *(const v4u*)__builtin_assume_aligned(p + 32, 16);
}
__device__ __forceinline__ void frag_ldB(FragBF& f, const char* p) {
    f.q[0] = *(const v4u*)__builtin_assume_aligned(p, 16);
    f.q[1] = *(const v4u*)__builtin_assume_aligned(p + 32, 16);
}

// ---- one-shot converts ------------------------------------------------------
__global__ void __launch_bounds__(256)
cvt_kernel(const float4* __restrict__ in, bf16* __restrict__ out, int n4) {
    int i = blockIdx.x * 256 + threadIdx.x;
    if (i >= n4) return;
    float4 v = in[i];
    union { bf16 h[4]; unsigned long long u; } pk;
    pk.h[0] = (bf16)v.x; pk.h[1] = (bf16)v.y; pk.h[2] = (bf16)v.z; pk.h[3] = (bf16)v.w;
    *(unsigned long long*)(out + (size_t)i * 4) = pk.u;
}

// out[n][k] = (bf16) in[k][n]
__global__ void __launch_bounds__(256)
transpose_cvt_kernel(const float* __restrict__ in, bf16* __restrict__ out, int K, int N) {
    __shared__ float tile[32][33];
    const int n0 = blockIdx.x * 32, k0 = blockIdx.y * 32;
    const int tx = threadIdx.x, ty = threadIdx.y;
#pragma unroll
    for (int r = 0; r < 32; r += 8)
        tile[ty + r][tx] = in[(size_t)(k0 + ty + r) * N + n0 + tx];
    __syncthreads();
#pragma unroll
    for (int r = 0; r < 32; r += 8)
        out[(size_t)(n0 + ty + r) * K + k0 + tx] = (bf16)tile[tx][ty + r];
}

// ---- shared double-buffered bf16 GEMM core (A[M][K], Bt[N][K]) --------------
__device__ __forceinline__ void gemm_core_bf16(
    const bf16* __restrict__ A, const bf16* __restrict__ Bt,
    int lda, int ldb, int K, int m0, int n0, char* smem, v8f acc[2][4])
{
    const int tid  = threadIdx.x;
    const int lane = tid & 31;
    const int wid  = tid >> 5;
    const int wm   = wid & 3;
    const int wn   = wid >> 2;
    const int half = lane >> 4;
    const int l16  = lane & 15;
    const unsigned LB = __builtin_amdgcn_groupstaticsize();

    v8f zero = {};
#pragma unroll
    for (int i = 0; i < 2; i++)
#pragma unroll
        for (int j = 0; j < 4; j++) acc[i][j] = zero;

    // chunk decomposition: 512 16B-chunks per tile, 2 per thread; row=c>>2 sub=c&3
    const int r0 = tid >> 2,          s0 = tid & 3;
    const int r1 = (tid + 256) >> 2;  // sub identical

    auto issue = [&](int kb, unsigned buf) {
        async_ld_b128(LB + buf + G_A_OFF + (unsigned)(r0 * 80 + s0 * 16),
                      A + (size_t)(m0 + r0) * lda + kb + s0 * 8);
        async_ld_b128(LB + buf + G_A_OFF + (unsigned)(r1 * 80 + s0 * 16),
                      A + (size_t)(m0 + r1) * lda + kb + s0 * 8);
        async_ld_b128(LB + buf + G_B_OFF + (unsigned)(r0 * 80 + s0 * 16),
                      Bt + (size_t)(n0 + r0) * ldb + kb + s0 * 8);
        async_ld_b128(LB + buf + G_B_OFF + (unsigned)(r1 * 80 + s0 * 16),
                      Bt + (size_t)(n0 + r1) * ldb + kb + s0 * 8);
    };

    issue(0, 0u);
    const int NT = K / BK;
    for (int t = 0; t < NT; t++) {
        const unsigned cur = (t & 1) ? G_BUF : 0u;
        const unsigned nxt = (t & 1) ? 0u : G_BUF;
        __syncthreads();                       // everyone done reading buf[nxt]
        if (t + 1 < NT) { issue((t + 1) * BK, nxt); WAIT_ASYNC(4); }
        else            { WAIT_ASYNC(0); }
        __syncthreads();                       // buf[cur] visible to all waves

        const char* sa = smem + cur + G_A_OFF;
        const char* sb = smem + cur + G_B_OFF;
        FragBF fa[2], fb[4];
#pragma unroll
        for (int mt = 0; mt < 2; mt++)
            frag_ldA(fa[mt], sa + (wm * 32 + mt * 16 + l16) * 80 + half * 16);
#pragma unroll
        for (int nt = 0; nt < 4; nt++)
            frag_ldB(fb[nt], sb + (wn * 64 + nt * 16 + l16) * 80 + half * 16);
#pragma unroll
        for (int mt = 0; mt < 2; mt++)
#pragma unroll
            for (int nt = 0; nt < 4; nt++)
                acc[mt][nt] = __builtin_amdgcn_wmma_f32_16x16x32_bf16(
                    false, fa[mt].v, false, fb[nt].v,
                    (short)0, acc[mt][nt], false, false);
    }
}

// ---- Kernel 1: qkv = x@W_qkv + b; scatter Q,K (B,H,G,D) and V transposed ----
__global__ void __launch_bounds__(256)
qkv_gemm_kernel(const bf16* __restrict__ xb, const bf16* __restrict__ Wqt,
                const float* __restrict__ bias,
                bf16* __restrict__ Qb, bf16* __restrict__ Kb, bf16* __restrict__ Vt)
{
    extern __shared__ char smem[];
    const int m0 = blockIdx.x * BM, n0 = blockIdx.y * BN;
    v8f acc[2][4];
    gemm_core_bf16(xb, Wqt, EMB, EMB, EMB, m0, n0, smem, acc);

    const int lane = threadIdx.x & 31, wid = threadIdx.x >> 5;
    const int wm = wid & 3, wn = wid >> 2, half = lane >> 4, l16 = lane & 15;
#pragma unroll
    for (int mt = 0; mt < 2; mt++)
#pragma unroll
        for (int nt = 0; nt < 4; nt++)
#pragma unroll
            for (int r = 0; r < 8; r++) {
                int M = m0 + wm * 32 + mt * 16 + half * 8 + r;
                int N = n0 + wn * 64 + nt * 16 + l16;
                float v = acc[mt][nt][r] + bias[N];
                int s = N % 3, hd = N / 3;
                int h = hd / HDIM, d = hd % HDIM;
                int b = M / SEQ, g = M % SEQ;
                size_t bhh = (size_t)b * HEADS + h;
                bf16 bv = (bf16)v;
                if (s == 0)      Qb[(bhh * SEQ + g) * HDIM + d] = bv;
                else if (s == 1) Kb[(bhh * SEQ + g) * HDIM + d] = bv;
                else             Vt[(bhh * HDIM + d) * SEQ + g] = bv;
            }
}

// ---- Kernel 2: flash attention, async double-buffered K/V tiles -------------
__global__ void __launch_bounds__(256)
attn_kernel(const bf16* __restrict__ Qb, const bf16* __restrict__ Kb,
            const bf16* __restrict__ Vtb, bf16* __restrict__ Ob)
{
    extern __shared__ char smem[];
    const int tid  = threadIdx.x;
    const int lane = tid & 31;
    const int wid  = tid >> 5;
    const int half = lane >> 4;
    const int l16  = lane & 15;
    const unsigned LB = __builtin_amdgcn_groupstaticsize();

    const int bh = blockIdx.x >> 4;
    const int qt = blockIdx.x & 15;
    const size_t base = (size_t)bh * SEQ * HDIM;
    const bf16* Qp  = Qb + base;
    const bf16* Kp  = Kb + base;
    const bf16* Vtp = Vtb + base;   // (D, G) layout per (b,h)

    const int qrow0 = qt * 128 + wid * 16;

    // persistent Q fragments (16 rows x 192)
    FragBF fq[6];
    {
        const bf16* q = Qp + (size_t)(qrow0 + l16) * HDIM + half * 8;
#pragma unroll
        for (int kc = 0; kc < 6; kc++)
#pragma unroll
            for (int j = 0; j < 4; j++) {
                fq[kc].w[j]     = *(const unsigned*)(q + kc * 32 + 2 * j);
                fq[kc].w[4 + j] = *(const unsigned*)(q + kc * 32 + 16 + 2 * j);
            }
    }

    // chunk decomposition (kt-invariant): K tile 768 chunks, Vt tile 768 chunks
    int kKey[3], kCh[3], vD[3], vS[3];
#pragma unroll
    for (int i = 0; i < 3; i++) {
        int c = tid + i * 256;
        kKey[i] = c / 24; kCh[i] = c % 24;
        vD[i] = c >> 2;   vS[i] = c & 3;
    }

    auto issueKV = [&](int kt, unsigned buf) {
#pragma unroll
        for (int i = 0; i < 3; i++)
            async_ld_b128(LB + buf + (unsigned)(kKey[i] * 400 + kCh[i] * 16),
                          Kp + (size_t)(kt + kKey[i]) * HDIM + kCh[i] * 8);
#pragma unroll
        for (int i = 0; i < 3; i++)
            async_ld_b128(LB + buf + ATT_V_OFF + (unsigned)(vD[i] * 80 + vS[i] * 16),
                          Vtp + (size_t)vD[i] * SEQ + kt + vS[i] * 8);
    };

    v8f zero = {};
    v8f acco[12];
#pragma unroll
    for (int i = 0; i < 12; i++) acco[i] = zero;
    float mst[8], lst[8];
#pragma unroll
    for (int r = 0; r < 8; r++) { mst[r] = -1e30f; lst[r] = 0.0f; }

    const float sc = 0.03608439182435161f;   // 1/sqrt(768)
    char* sPb = smem + ATT_P_OFF + wid * 1280;

    issueKV(0, 0u);
    const int NT = SEQ / KT;   // 64
    for (int t = 0; t < NT; t++) {
        const unsigned cur = (t & 1) ? ATT_BUF : 0u;
        const unsigned nxt = (t & 1) ? 0u : ATT_BUF;
        __syncthreads();
        if (t + 1 < NT) { issueKV((t + 1) * KT, nxt); WAIT_ASYNC(6); }
        else            { WAIT_ASYNC(0); }
        __syncthreads();

        const char* sKb = smem + cur;               // [key][d], stride 400B
        const char* sVb = smem + cur + ATT_V_OFF;   // [d][key], stride 80B

        // S = Q @ K^T (16 x 32)
        v8f accs[2] = { zero, zero };
#pragma unroll
        for (int kc = 0; kc < 6; kc++)
#pragma unroll
            for (int tt = 0; tt < 2; tt++) {
                FragBF fb;
                frag_ldB(fb, sKb + (tt * 16 + l16) * 400 + kc * 64 + half * 16);
                accs[tt] = __builtin_amdgcn_wmma_f32_16x16x32_bf16(
                    false, fq[kc].v, false, fb.v, (short)0, accs[tt], false, false);
            }

        // online softmax, P -> per-wave LDS scratch
#pragma unroll
        for (int r = 0; r < 8; r++) {
            float x0 = accs[0][r] * sc;
            float x1 = accs[1][r] * sc;
            float mx = fmaxf(x0, x1);
#pragma unroll
            for (int m = 1; m < 16; m <<= 1) mx = fmaxf(mx, __shfl_xor(mx, m, 32));
            float mn    = fmaxf(mst[r], mx);
            float alpha = __expf(mst[r] - mn);
            float p0 = __expf(x0 - mn);
            float p1 = __expf(x1 - mn);
            float rs = p0 + p1;
#pragma unroll
            for (int m = 1; m < 16; m <<= 1) rs += __shfl_xor(rs, m, 32);
            lst[r] = lst[r] * alpha + rs;
            mst[r] = mn;
#pragma unroll
            for (int ot = 0; ot < 12; ot++) acco[ot][r] = acco[ot][r] * alpha;
            int prow = half * 8 + r;
            *(bf16*)(sPb + prow * 80 + l16 * 2)      = (bf16)p0;
            *(bf16*)(sPb + prow * 80 + 32 + l16 * 2) = (bf16)p1;
        }
        WAIT_DS0();   // wave-private LDS RAW before fragment reload

        FragBF fp;
        frag_ldA(fp, sPb + l16 * 80 + half * 16);

        // O += P @ V (16 x 192)
#pragma unroll
        for (int ot = 0; ot < 12; ot++) {
            FragBF fv;
            frag_ldB(fv, sVb + (ot * 16 + l16) * 80 + half * 16);
            acco[ot] = __builtin_amdgcn_wmma_f32_16x16x32_bf16(
                false, fp.v, false, fv.v, (short)0, acco[ot], false, false);
        }
    }

    const int b = bh >> 2;
    const int h = bh & 3;
#pragma unroll
    for (int ot = 0; ot < 12; ot++)
#pragma unroll
        for (int r = 0; r < 8; r++) {
            float v = acco[ot][r] / lst[r];
            int g = qrow0 + half * 8 + r;
            int d = ot * 16 + l16;
            Ob[((size_t)b * SEQ + g) * EMB + h * HDIM + d] = (bf16)v;
        }
}

// ---- Kernel 3: out = O @ W_proj + b_proj (fp32 out) -------------------------
__global__ void __launch_bounds__(256)
proj_gemm_kernel(const bf16* __restrict__ Ob, const bf16* __restrict__ Wpt,
                 const float* __restrict__ bias, float* __restrict__ out)
{
    extern __shared__ char smem[];
    const int m0 = blockIdx.x * BM, n0 = blockIdx.y * BN;
    v8f acc[2][4];
    gemm_core_bf16(Ob, Wpt, EMB, EMB, EMB, m0, n0, smem, acc);

    const int lane = threadIdx.x & 31, wid = threadIdx.x >> 5;
    const int wm = wid & 3, wn = wid >> 2, half = lane >> 4, l16 = lane & 15;
#pragma unroll
    for (int mt = 0; mt < 2; mt++)
#pragma unroll
        for (int nt = 0; nt < 4; nt++)
#pragma unroll
            for (int r = 0; r < 8; r++) {
                int M = m0 + wm * 32 + mt * 16 + half * 8 + r;
                int N = n0 + wn * 64 + nt * 16 + l16;
                out[(size_t)M * EMB + N] = acc[mt][nt][r] + bias[N];
            }
}

// ---------------------------------------------------------------------------
extern "C" void kernel_launch(void* const* d_in, const int* in_sizes, int n_in,
                              void* d_out, int out_size, void* d_ws, size_t ws_size,
                              hipStream_t stream) {
    (void)in_sizes; (void)n_in; (void)out_size; (void)ws_size;
    const float* x      = (const float*)d_in[0];
    const float* W_qkv  = (const float*)d_in[1];
    const float* b_qkv  = (const float*)d_in[2];
    const float* W_proj = (const float*)d_in[3];
    const float* b_proj = (const float*)d_in[4];
    float* out = (float*)d_out;

    const size_t nBH = (size_t)BATCH * HEADS * SEQ * HDIM;  // 6.29M
    bf16* Qb  = (bf16*)d_ws;
    bf16* Kb  = Qb + nBH;
    bf16* Vt  = Kb + nBH;                    // V stored (B,H,D,G)
    bf16* Obf = Vt + nBH;
    bf16* xb  = Obf + nBH;
    bf16* Wqt = xb + (size_t)MROWS * EMB;    // W_qkv^T  [2304][768]
    bf16* Wpt = Wqt + (size_t)NQKV * EMB;    // W_proj^T [768][768]
    // total workspace ~67.6 MB bf16

    cvt_kernel<<<(MROWS * EMB / 4 + 255) / 256, 256, 0, stream>>>(
        (const float4*)x, xb, MROWS * EMB / 4);
    transpose_cvt_kernel<<<dim3(NQKV / 32, EMB / 32), dim3(32, 8), 0, stream>>>(
        W_qkv, Wqt, EMB, NQKV);
    transpose_cvt_kernel<<<dim3(EMB / 32, EMB / 32), dim3(32, 8), 0, stream>>>(
        W_proj, Wpt, EMB, EMB);

    qkv_gemm_kernel<<<dim3(MROWS / BM, NQKV / BN), 256, 40960, stream>>>(
        xb, Wqt, b_qkv, Qb, Kb, Vt);

    attn_kernel<<<BATCH * HEADS * (SEQ / 128), 256, 66560, stream>>>(
        Qb, Kb, Vt, Obf);

    proj_gemm_kernel<<<dim3(MROWS / BM, EMB / BN), 256, 40960, stream>>>(
        Obf, Wpt, b_proj, out);
}
